// StandardAttention_5815385719067
// MI455X (gfx1250) — matrix operand order repaired and checked by reading, not compiled, not run on hardware
//
#include <hip/hip_runtime.h>
#include <hip/hip_bf16.h>

// Multi-head attention, B=2, N=2048, D=1024, H=16, DH=64, fp32 in/out.
// Pipeline (all f16 WMMA w/ fp32 accum):
//  1) convert x, Wq, Wk, Wv, Wo -> f16 in workspace
//  2) Q = x@Wq^T+bq, K = x@Wk^T+bk   (f16, layout [b,h,n,dh])
//     V = x@Wv^T+bv                  (f16, TRANSPOSED layout [b,h,dh,n])
//  3) flash attention per 16-query tile per wave:
//     S^T = K_tile * Q^T (lane = query col), online softmax, O^T = V^T * P^T
//     -> attn_out f16 [b*n, h*dh]
//  4) out = attn_out @ Wo^T + bo (fp32)
// Workspace usage: 48 MB.

typedef _Float16 h8  __attribute__((ext_vector_type(8)));
typedef _Float16 h16 __attribute__((ext_vector_type(16)));
typedef float    v8f __attribute__((ext_vector_type(8)));

#define BATCH 2
#define SEQ   2048
#define DMODEL 1024
#define HEADS 16
#define DHEAD 64
#define ROWS  (BATCH*SEQ)   // 4096

__device__ __forceinline__ v8f wmma32(h16 a, h16 b, v8f c) {
  return __builtin_amdgcn_wmma_f32_16x16x32_f16(false, a, false, b, (short)0, c, false, false);
}

// A-matrix 16x32 f16 fragment (M x K), row-major source.
// lane L holds row (rowbase + L%16); K layout (contiguous, matches B):
//   lanes 0-15:  K = k0+0..15
//   lanes 16-31: K = k0+16..31
__device__ __forceinline__ h16 load_a_frag(const _Float16* __restrict__ base,
                                           int ld, int rowbase, int k0, int lane) {
  int r   = rowbase + (lane & 15);
  int sel = (lane >> 4) & 1;
  const _Float16* p = base + (size_t)r * ld + k0 + sel * 16;
  union { h16 v; h8 h[2]; } u;
  u.h[0] = *(const h8*)(p);
  u.h[1] = *(const h8*)(p + 8);
  return u.v;
}

// B-matrix 32x16 f16 fragment (K x N); B[k][n] read from row-major rows of the
// "n" operand (W row n, or q/key row n). Contiguous K order (same as load_a_frag):
//   lanes 0-15:  col n = L,    K = k0+0..15
//   lanes 16-31: col n = L-16, K = k0+16..31
__device__ __forceinline__ h16 load_b_frag(const _Float16* __restrict__ base,
                                           int ld, int colbase, int k0, int lane) {
  int r   = colbase + (lane & 15);
  int sel = (lane >> 4) & 1;
  const _Float16* p = base + (size_t)r * ld + k0 + sel * 16;
  union { h16 v; h8 h[2]; } u;
  u.h[0] = *(const h8*)(p);
  u.h[1] = *(const h8*)(p + 8);
  return u.v;
}

__global__ void cvt_f32_f16(const float* __restrict__ src, _Float16* __restrict__ dst, int n) {
  int i = blockIdx.x * blockDim.x + threadIdx.x;
  int stride = gridDim.x * blockDim.x;
  for (; i < n; i += stride) dst[i] = (_Float16)src[i];
}

// C = A[4096x1024] * Bw^T + bias.  Block = 8 waves, wave tile 32x64, WG tile 128x128.
// MODE 1: f16 out, [b,h,n,dh] layout (Q and K)
// MODE 3: f16 out, [b,h,dh,n] layout (V transposed)
// MODE 4: f32 out, row-major [row, col] (final projection)
template <int MODE>
__global__ __launch_bounds__(256) void gemm_f16(const _Float16* __restrict__ A,
                                                const _Float16* __restrict__ Bw,
                                                const float* __restrict__ bias,
                                                void* __restrict__ outp) {
  const int lane = threadIdx.x & 31;
  const int w = threadIdx.x >> 5;
  const int wm = w & 3, wn = w >> 2;
  const int rowbase = blockIdx.x * 128 + wm * 32;
  const int colbase = blockIdx.y * 128 + wn * 64;

  v8f acc[2][4];
  v8f z = {};
  for (int t = 0; t < 2; ++t)
    for (int u = 0; u < 4; ++u) acc[t][u] = z;

  for (int k0 = 0; k0 < DMODEL; k0 += 32) {
    h16 af[2], bf[4];
#pragma unroll
    for (int t = 0; t < 2; ++t) af[t] = load_a_frag(A, DMODEL, rowbase + t * 16, k0, lane);
#pragma unroll
    for (int u = 0; u < 4; ++u) bf[u] = load_b_frag(Bw, DMODEL, colbase + u * 16, k0, lane);
#pragma unroll
    for (int t = 0; t < 2; ++t)
#pragma unroll
      for (int u = 0; u < 4; ++u) acc[t][u] = wmma32(af[t], bf[u], acc[t][u]);
  }

  const int sel = lane >> 4;
  const int cl = lane & 15;
#pragma unroll
  for (int t = 0; t < 2; ++t) {
#pragma unroll
    for (int u = 0; u < 4; ++u) {
      const int col = colbase + u * 16 + cl;          // output feature
      const float bv = bias[col];
      const int row0 = rowbase + t * 16 + sel * 8;    // first of 8 consecutive rows
      if constexpr (MODE == 3) {
        // vT[b][h][dh][n]: 8 consecutive n per lane -> one 16B store
        const int b = row0 >> 11, n0 = row0 & (SEQ - 1);
        const int h = col >> 6, dh = col & 63;
        h8 pk;
#pragma unroll
        for (int i = 0; i < 8; ++i) pk[i] = (_Float16)(acc[t][u][i] + bv);
        _Float16* dst = (_Float16*)outp +
            (((size_t)(b * HEADS + h) * DHEAD + dh) * SEQ + n0);
        *(h8*)dst = pk;
      } else if constexpr (MODE == 4) {
        float* dst = (float*)outp;
#pragma unroll
        for (int i = 0; i < 8; ++i)
          dst[(size_t)(row0 + i) * DMODEL + col] = acc[t][u][i] + bv;
      } else {
        // q/k[b][h][n][dh]
        _Float16* dst = (_Float16*)outp;
        const int h = col >> 6, dh = col & 63;
#pragma unroll
        for (int i = 0; i < 8; ++i) {
          const int row = row0 + i;
          const int b = row >> 11, n = row & (SEQ - 1);
          dst[(((size_t)(b * HEADS + h)) * SEQ + n) * DHEAD + dh] = (_Float16)(acc[t][u][i] + bv);
        }
      }
    }
  }
}

// Flash attention: each wave handles 16 queries of one (b,h), streaming keys
// in chunks of 32. grid = 512 blocks x 8 waves = 4096 query tiles.
__global__ __launch_bounds__(256) void attn_kernel(const _Float16* __restrict__ q,
                                                   const _Float16* __restrict__ k,
                                                   const _Float16* __restrict__ vt,
                                                   _Float16* __restrict__ ao) {
  const int lane = threadIdx.x & 31;
  const int w = threadIdx.x >> 5;
  const int gw = blockIdx.x * 8 + w;      // 0..4095
  const int qt = gw & 127;                // query tile in sequence
  const int bh = gw >> 7;                 // 0..31  (b*16 + h)
  const int qbase = qt * 16;

  const _Float16* qp = q + ((size_t)bh * SEQ + qbase) * DHEAD;
  const _Float16* kp = k + (size_t)bh * SEQ * DHEAD;
  const _Float16* vp = vt + (size_t)bh * DHEAD * SEQ;

  // Q^T as B-fragments (K-dim = dh, split 0..31 / 32..63); lane = query col.
  const h16 bq0 = load_b_frag(qp, DHEAD, 0, 0, lane);
  const h16 bq1 = load_b_frag(qp, DHEAD, 0, 32, lane);

  const float csc = 0.18033688011112042f; // (1/sqrt(64)) * log2(e)
  float m = -1.0e30f, lsum = 0.0f;
  v8f acc[4];
  v8f z = {};
  for (int d = 0; d < 4; ++d) acc[d] = z;

  for (int kb = 0; kb < SEQ; kb += 32) {
    // S^T tiles: rows = keys (vgpr), cols = queries (lane)
    h16 ka00 = load_a_frag(kp, DHEAD, kb, 0, lane);
    h16 ka01 = load_a_frag(kp, DHEAD, kb, 32, lane);
    h16 ka10 = load_a_frag(kp, DHEAD, kb + 16, 0, lane);
    h16 ka11 = load_a_frag(kp, DHEAD, kb + 16, 32, lane);
    v8f c0 = z; c0 = wmma32(ka00, bq0, c0); c0 = wmma32(ka01, bq1, c0);
    v8f c1 = z; c1 = wmma32(ka10, bq0, c1); c1 = wmma32(ka11, bq1, c1);

    // per-query (per-lane) running max over 32 keys of this chunk
    float mloc = c0[0];
#pragma unroll
    for (int i = 1; i < 8; ++i) mloc = fmaxf(mloc, c0[i]);
#pragma unroll
    for (int i = 0; i < 8; ++i) mloc = fmaxf(mloc, c1[i]);
    mloc = fmaxf(mloc, __shfl_xor(mloc, 16, 32));
    const float mnew = fmaxf(m, mloc);
    const float alpha = __builtin_amdgcn_exp2f((m - mnew) * csc);
    m = mnew;

    float e0[8], e1[8];
    float s = 0.0f;
#pragma unroll
    for (int i = 0; i < 8; ++i) {
      e0[i] = __builtin_amdgcn_exp2f((c0[i] - m) * csc); s += e0[i];
      e1[i] = __builtin_amdgcn_exp2f((c1[i] - m) * csc); s += e1[i];
    }
    s += __shfl_xor(s, 16, 32);
    lsum = lsum * alpha + s;
#pragma unroll
    for (int d = 0; d < 4; ++d)
#pragma unroll
      for (int i = 0; i < 8; ++i) acc[d][i] *= alpha;

    // Assemble P^T B-fragment (K=32 keys x 16 queries):
    //   lanes 0-15 need keys kb+0..15  = {own e0 (kb+0..7), e0 from lane+16 (kb+8..15)}
    //   lanes16-31 need keys kb+16..31 = {e1 from lane-16 (kb+16..23), own e1 (kb+24..31)}
    const int hi = lane >> 4;
    h16 pb;
#pragma unroll
    for (int i = 0; i < 8; ++i) {
      const float o0 = __shfl_xor(e0[i], 16, 32);
      const float o1 = __shfl_xor(e1[i], 16, 32);
      pb[i]     = (_Float16)(hi ? o1 : e0[i]);
      pb[i + 8] = (_Float16)(hi ? e1[i] : o0);
    }

    // O^T += V^T * P^T ; rows = dh (4 tiles), cols = queries
#pragma unroll
    for (int d = 0; d < 4; ++d) {
      h16 va = load_a_frag(vp, SEQ, d * 16, kb, lane);
      acc[d] = wmma32(va, pb, acc[d]);
    }
  }

  const float inv = 1.0f / lsum;
  const int b = bh >> 4, h = bh & 15;
  const size_t row = (size_t)(b * SEQ + qbase + (lane & 15)); // query row in [4096]
  const int coff = h * DHEAD + (lane >> 4) * 8;
#pragma unroll
  for (int d = 0; d < 4; ++d) {
    h8 pk;
#pragma unroll
    for (int i = 0; i < 8; ++i) pk[i] = (_Float16)(acc[d][i] * inv);
    *(h8*)(ao + row * DMODEL + coff + d * 16) = pk;
  }
}

extern "C" void kernel_launch(void* const* d_in, const int* in_sizes, int n_in,
                              void* d_out, int out_size, void* d_ws, size_t ws_size,
                              hipStream_t stream) {
  const float* x  = (const float*)d_in[0];
  const float* Wq = (const float*)d_in[1];
  const float* bq = (const float*)d_in[2];
  const float* Wk = (const float*)d_in[3];
  const float* bk = (const float*)d_in[4];
  const float* Wv = (const float*)d_in[5];
  const float* bv = (const float*)d_in[6];
  const float* Wo = (const float*)d_in[7];
  const float* bo = (const float*)d_in[8];

  char* ws = (char*)d_ws;
  const size_t MB = 1u << 20;
  _Float16* xh  = (_Float16*)(ws + 0);        // 8 MB  (4096x1024)
  _Float16* wqh = (_Float16*)(ws + 8 * MB);   // 2 MB
  _Float16* wkh = (_Float16*)(ws + 10 * MB);
  _Float16* wvh = (_Float16*)(ws + 12 * MB);
  _Float16* woh = (_Float16*)(ws + 14 * MB);
  _Float16* qb  = (_Float16*)(ws + 16 * MB);  // 8 MB [b,h,n,dh]
  _Float16* kb_ = (_Float16*)(ws + 24 * MB);  // 8 MB [b,h,n,dh]
  _Float16* vtb = (_Float16*)(ws + 32 * MB);  // 8 MB [b,h,dh,n]
  _Float16* aob = (_Float16*)(ws + 40 * MB);  // 8 MB [4096,1024]

  const int nx = ROWS * DMODEL;       // 4,194,304
  const int nw = DMODEL * DMODEL;     // 1,048,576
  cvt_f32_f16<<<2048, 256, 0, stream>>>(x, xh, nx);
  cvt_f32_f16<<<1024, 256, 0, stream>>>(Wq, wqh, nw);
  cvt_f32_f16<<<1024, 256, 0, stream>>>(Wk, wkh, nw);
  cvt_f32_f16<<<1024, 256, 0, stream>>>(Wv, wvh, nw);
  cvt_f32_f16<<<1024, 256, 0, stream>>>(Wo, woh, nw);

  dim3 gg(ROWS / 128, DMODEL / 128);  // (32, 8)
  gemm_f16<1><<<gg, 256, 0, stream>>>(xh, wqh, bq, (void*)qb);
  gemm_f16<1><<<gg, 256, 0, stream>>>(xh, wkh, bk, (void*)kb_);
  gemm_f16<3><<<gg, 256, 0, stream>>>(xh, wvh, bv, (void*)vtb);

  attn_kernel<<<512, 256, 0, stream>>>(qb, kb_, vtb, aob);

  gemm_f16<4><<<gg, 256, 0, stream>>>(aob, woh, bo, d_out);
}
